// Encoder_27127013441931
// MI455X (gfx1250) — matrix operand order repaired and checked by reading, not compiled, not run on hardware
//
#include <hip/hip_runtime.h>
#include <hip/hip_bf16.h>
#include <stdint.h>

// ---------------------------------------------------------------------------
// Problem constants (from reference setup_inputs)
// ---------------------------------------------------------------------------
#define BB 8
#define TT 768
#define CC 512
#define FF 2048
#define HH 8
#define KC 64
#define NL 6
#define MROWS (BB * TT)   // 6144

typedef _Float16 half_t;
typedef __attribute__((ext_vector_type(16))) _Float16 v16h;
typedef __attribute__((ext_vector_type(8)))  _Float16 v8h;
typedef __attribute__((ext_vector_type(8)))  float    v8f;

// Build a v16h WMMA fragment from two aligned contiguous 8-half vectors.
__device__ __forceinline__ v16h frag16(const half_t* p0, const half_t* p1) {
    v8h lo = *(const v8h*)p0;
    v8h hi = *(const v8h*)p1;
    return __builtin_shufflevector(lo, hi, 0, 1, 2, 3, 4, 5, 6, 7,
                                          8, 9, 10, 11, 12, 13, 14, 15);
}

__device__ __forceinline__ v8f wmma(v16h a, v16h b, v8f c) {
    return __builtin_amdgcn_wmma_f32_16x16x32_f16(false, a, false, b,
                                                  (short)0, c, false, false);
}

// ---------------------------------------------------------------------------
// f32 [K,N] -> f16 [N,K] transpose-convert (per layer in blockIdx.z)
// ---------------------------------------------------------------------------
__global__ __launch_bounds__(256) void k_cvt_t(const float* __restrict__ src,
                                               half_t* __restrict__ dst,
                                               int K, int N) {
    __shared__ half_t tile[32][33];
    const size_t lo = (size_t)blockIdx.z * K * N;
    const float* s = src + lo;
    half_t* d = dst + lo;
    const int n0 = blockIdx.x * 32, k0 = blockIdx.y * 32;
    const int c = threadIdx.x & 31, r = threadIdx.x >> 5;
#pragma unroll
    for (int rr = r; rr < 32; rr += 8)
        tile[rr][c] = (half_t)s[(size_t)(k0 + rr) * N + n0 + c];
    __syncthreads();
#pragma unroll
    for (int rr = r; rr < 32; rr += 8)
        d[(size_t)(n0 + rr) * K + k0 + c] = tile[c][rr];
}

// x = x_in * mask_row ; also emit f16 copy
__global__ void k_premask(const float* __restrict__ xin, const float* __restrict__ mask,
                          float* __restrict__ x, half_t* __restrict__ xh) {
    int i = blockIdx.x * 256 + threadIdx.x;      // exact grid: MROWS*CC
    int row = i / CC;
    float v = xin[i] * mask[row];
    x[i] = v;
    xh[i] = (half_t)v;
}

// d_out = x * mask_row
__global__ void k_final(const float* __restrict__ x, const float* __restrict__ mask,
                        float* __restrict__ out) {
    int i = blockIdx.x * 256 + threadIdx.x;
    out[i] = x[i] * mask[i / CC];
}

// ---------------------------------------------------------------------------
// WMMA GEMM:  D[M,N] = act( A[M,K] * Bt[N,K]^T + bias ) * oscale * rowmask
// A f16 [M,K]; Bt f16 [N,K] (pre-transposed weight).  Tile 128x128, 8 waves.
// Double-buffered LDS pipeline: stage chunk c+1 while computing chunk c.
// sched_group_barrier batches the 18 fragment ds_loads ahead of the 8 WMMAs
// so the wait-inserter emits partial dscnt waits (load/XDL overlap).
// ---------------------------------------------------------------------------
__global__ __launch_bounds__(256) void k_gemm(
    const half_t* __restrict__ A, const half_t* __restrict__ Bt,
    const float* __restrict__ bias, const float* __restrict__ rowmask,
    float* __restrict__ Cf, half_t* __restrict__ Ch,
    int M, int N, int K, int relu, float oscale)
{
    __shared__ half_t As[2][128][48];   // 96B rows: 16B/32B aligned vector access
    __shared__ half_t Bs[2][128][48];

    const int tid  = threadIdx.x;
    const int lane = tid & 31;
    const int w    = tid >> 5;
    const int n0   = blockIdx.x * 128;
    const int m0   = blockIdx.y * 128;
    const int nchunk = K >> 5;

    const int r0 = tid >> 2,           c0 = (tid & 3) * 8;          // vec 0
    const int r1 = (tid + 256) >> 2,   c1 = (tid & 3) * 8;          // vec 1

    // load phase then store phase: keeps 4 global loads in flight together
    auto stage = [&](int buf, int k0) {
        uint4 la0 = *(const uint4*)&A [(size_t)(m0 + r0) * K + k0 + c0];
        uint4 la1 = *(const uint4*)&A [(size_t)(m0 + r1) * K + k0 + c1];
        uint4 lb0 = *(const uint4*)&Bt[(size_t)(n0 + r0) * K + k0 + c0];
        uint4 lb1 = *(const uint4*)&Bt[(size_t)(n0 + r1) * K + k0 + c1];
        *(uint4*)&As[buf][r0][c0] = la0;
        *(uint4*)&As[buf][r1][c1] = la1;
        *(uint4*)&Bs[buf][r0][c0] = lb0;
        *(uint4*)&Bs[buf][r1][c1] = lb1;
    };

    v8f acc[8] = {};
    stage(0, 0);

    const int arow = w * 16 + (lane & 15);
    const int hh   = (lane >> 4) << 3;            // 0 or 8
    const int kh   = (lane >> 4) << 4;            // 0 or 16
    const int crow = lane & 15;

    for (int c = 0; c < nchunk; ++c) {
        __syncthreads();                           // buffer c&1 ready
        if (c + 2 < nchunk) {                      // prefetch 2 chunks ahead
            __builtin_prefetch(&A [(size_t)(m0 + (tid >> 1)) * K + (c + 2) * 32], 0, 3);
            __builtin_prefetch(&Bt[(size_t)(n0 + (tid >> 1)) * K + (c + 2) * 32], 0, 3);
        }
        if (c + 1 < nchunk) stage((c + 1) & 1, (c + 1) * 32);

        const int buf = c & 1;
        v16h a = frag16(&As[buf][arow][kh], &As[buf][arow][kh + 8]);
        v16h bf[8];
#pragma unroll
        for (int ct = 0; ct < 8; ++ct)
            bf[ct] = *(const v16h*)&Bs[buf][ct * 16 + crow][kh];
#pragma unroll
        for (int ct = 0; ct < 8; ++ct)
            acc[ct] = wmma(a, bf[ct], acc[ct]);

#if __has_builtin(__builtin_amdgcn_sched_group_barrier)
        // 18 DS reads (A frag: 2, B frags: 16), then 8 WMMAs back-to-back
        __builtin_amdgcn_sched_group_barrier(0x100, 18, 0);
        __builtin_amdgcn_sched_group_barrier(0x008, 8, 0);
#endif
    }

    // epilogue (C/D layout: lanes 0-15 -> M=r, lanes 16-31 -> M=r+8)
#pragma unroll
    for (int ct = 0; ct < 8; ++ct) {
        const int col = n0 + ct * 16 + (lane & 15);
        const float bv = bias ? bias[col] : 0.0f;
#pragma unroll
        for (int r = 0; r < 8; ++r) {
            const int row = m0 + w * 16 + r + ((lane >> 4) << 3);
            float val = acc[ct][r] + bv;
            if (relu) val = fmaxf(val, 0.0f);
            val *= oscale;
            if (rowmask) val *= rowmask[row];
            if (Cf) Cf[(size_t)row * N + col] = val;
            if (Ch) Ch[(size_t)row * N + col] = (half_t)val;
        }
    }
}

// ---------------------------------------------------------------------------
// Fused relative-position attention for one (b, h, 32-query tile).
// q pre-scaled by 1/sqrt(KC).  Band rel-k / rel-v (|j-i|<=4) folded in.
// Both GEMM phases double-buffer their LDS key/value tiles.
// ---------------------------------------------------------------------------
__global__ __launch_bounds__(256) void k_attn(
    const half_t* __restrict__ qh, const half_t* __restrict__ kh,
    const half_t* __restrict__ vh, const float* __restrict__ mask,
    const float* __restrict__ rk, const float* __restrict__ rv,
    half_t* __restrict__ out)
{
    constexpr int QT = 32, SLD = 776, KLD = 72, NG = TT / 64;
    __shared__ float  s[QT][SLD];      // score strip, then softmax probs (f32)
    __shared__ half_t qs[QT][KLD];     // q tile  (rows=query, cols=channel)
    __shared__ half_t kv[2][64][KLD];  // k tiles [key][chan]; later v^T [chan][key]
    __shared__ float  ot[QT][KLD];     // p.V output tile

    const int tid  = threadIdx.x;
    const int lane = tid & 31;
    const int w    = tid >> 5;
    const int q0   = blockIdx.x * QT;
    const int h    = blockIdx.y;
    const int b    = blockIdx.z;
    const size_t base = ((size_t)b * TT) * CC + (size_t)h * KC;  // + t*CC + d

    const int qt = w >> 2, kt = w & 3;           // wave's 16x16 tile per key group
    const int arow = qt * 16 + (lane & 15);
    const int col  = kt * 16 + (lane & 15);
    const int hh   = (lane >> 4) << 3;
    const int kh2  = (lane >> 4) << 4;

    const int sr0 = tid >> 3,         sc0 = (tid & 7) * 8;       // vec 0
    const int sr1 = (tid + 256) >> 3, sc1 = (tid & 7) * 8;       // vec 1

    auto stage_k = [&](int buf, int g) {
        uint4 l0 = *(const uint4*)&kh[base + (size_t)(g * 64 + sr0) * CC + sc0];
        uint4 l1 = *(const uint4*)&kh[base + (size_t)(g * 64 + sr1) * CC + sc1];
        *(uint4*)&kv[buf][sr0][sc0] = l0;
        *(uint4*)&kv[buf][sr1][sc1] = l1;
    };
    auto stage_vt = [&](int buf, int g) {        // transposed: kv = v^T [chan][key]
        v8h l0 = *(const v8h*)&vh[base + (size_t)(g * 64 + sr0) * CC + sc0];
        v8h l1 = *(const v8h*)&vh[base + (size_t)(g * 64 + sr1) * CC + sc1];
#pragma unroll
        for (int m = 0; m < 8; ++m) kv[buf][sc0 + m][sr0] = l0[m];
#pragma unroll
        for (int m = 0; m < 8; ++m) kv[buf][sc1 + m][sr1] = l1[m];
    };

    // ---- load q tile (32x64 halfs) + first k tile ----
    {
        int r = tid >> 3, c = (tid & 7) * 8;
        *(uint4*)&qs[r][c] = *(const uint4*)&qh[base + (size_t)(q0 + r) * CC + c];
    }
    stage_k(0, 0);
    __syncthreads();

    // q fragments are loop-invariant: hoist
    const v16h qa0 = frag16(&qs[arow][hh],      &qs[arow][16 + hh]);
    const v16h qa1 = frag16(&qs[arow][32 + hh], &qs[arow][48 + hh]);

    // ---- scores = q . k^T (reduction along contiguous channel axis) ----
    for (int g = 0; g < NG; ++g) {
        if (g) __syncthreads();
        if (g + 1 < NG) stage_k((g + 1) & 1, g + 1);
        const int buf = g & 1;
        v16h b0 = frag16(&kv[buf][col][hh],      &kv[buf][col][16 + hh]);
        v16h b1 = frag16(&kv[buf][col][32 + hh], &kv[buf][col][48 + hh]);
        v8f acc = {};
        acc = wmma(qa0, b0, acc);
        acc = wmma(qa1, b1, acc);
#pragma unroll
        for (int r = 0; r < 8; ++r)
            s[qt * 16 + r + ((lane >> 4) << 3)][g * 64 + kt * 16 + (lane & 15)] = acc[r];
    }
    __syncthreads();

    // ---- banded rel-k logits: s[i][i+dl-4] += q_scaled[i] . rk[dl] ----
    for (int t = tid; t < QT * 9; t += 256) {
        int i = t / 9, dl = t % 9;
        int j = q0 + i + dl - 4;
        if (j >= 0 && j < TT) {
            float a2 = 0.0f;
            for (int d = 0; d < KC; ++d) a2 += (float)qs[i][d] * rk[dl * KC + d];
            s[i][j] += a2;
        }
    }
    __syncthreads();

    // ---- mask ----
    for (int e = tid; e < QT * TT; e += 256) {
        int i = e / TT, j = e % TT;
        if (mask[b * TT + q0 + i] * mask[b * TT + j] == 0.0f) s[i][j] = -10000.0f;
    }
    __syncthreads();

    // ---- softmax: 4 rows per wave, lane-parallel over 768 cols ----
    for (int i = w * 4; i < w * 4 + 4; ++i) {
        float mx = -3.4e38f;
        for (int j = lane; j < TT; j += 32) mx = fmaxf(mx, s[i][j]);
#pragma unroll
        for (int o = 16; o; o >>= 1) mx = fmaxf(mx, __shfl_xor(mx, o, 32));
        float sum = 0.0f;
        for (int j = lane; j < TT; j += 32) {
            float p = __expf(s[i][j] - mx);
            s[i][j] = p;
            sum += p;
        }
#pragma unroll
        for (int o = 16; o; o >>= 1) sum += __shfl_xor(sum, o, 32);
        float inv = 1.0f / sum;
        for (int j = lane; j < TT; j += 32) s[i][j] *= inv;
    }
    __syncthreads();

    // ---- out = p . V : V staged transposed so reduction (key) is contiguous ----
    stage_vt(0, 0);
    v8f oacc = {};
    for (int g = 0; g < NG; ++g) {
        __syncthreads();
        if (g + 1 < NG) stage_vt((g + 1) & 1, g + 1);
        const int buf = g & 1;
        // p fragments: contiguous f32 runs -> f16
        const float* sp = &s[arow][g * 64];
        v16h a0, a1;
#pragma unroll
        for (int m = 0; m < 8; ++m) {
            a0[m]     = (half_t)sp[hh + m];
            a0[8 + m] = (half_t)sp[16 + hh + m];
            a1[m]     = (half_t)sp[32 + hh + m];
            a1[8 + m] = (half_t)sp[48 + hh + m];
        }
        v16h b0 = frag16(&kv[buf][col][hh],      &kv[buf][col][16 + hh]);
        v16h b1 = frag16(&kv[buf][col][32 + hh], &kv[buf][col][48 + hh]);
        oacc = wmma(a0, b0, oacc);
        oacc = wmma(a1, b1, oacc);
    }
#pragma unroll
    for (int r = 0; r < 8; ++r)
        ot[qt * 16 + r + ((lane >> 4) << 3)][kt * 16 + (lane & 15)] = oacc[r];
    __syncthreads();

    // ---- banded rel-v: out[i][d] += sum_dl p[i][i+dl-4] * rv[dl][d]; store f16 ----
    for (int e = tid; e < QT * KC; e += 256) {
        int i = e >> 6, d = e & 63;
        float a2 = ot[i][d];
#pragma unroll
        for (int dl = 0; dl < 9; ++dl) {
            int j = q0 + i + dl - 4;
            if (j >= 0 && j < TT) a2 += s[i][j] * rv[dl * KC + d];
        }
        out[base + (size_t)(q0 + i) * CC + d] = (half_t)a2;
    }
}

// ---------------------------------------------------------------------------
// x_out = LayerNorm(x_in + y) * gamma + beta ;  hout = f16(x_out * optional mask)
// ---------------------------------------------------------------------------
__global__ __launch_bounds__(256) void k_add_ln(
    const float* __restrict__ xin, const float* __restrict__ y,
    const float* __restrict__ gamma, const float* __restrict__ beta,
    const float* __restrict__ rowmask,
    float* __restrict__ xout, half_t* __restrict__ hout)
{
    __shared__ float red[8];
    const int row = blockIdx.x;
    const int tid = threadIdx.x, lane = tid & 31, w = tid >> 5;
    const size_t base = (size_t)row * CC;

    float v0 = xin[base + tid]       + y[base + tid];
    float v1 = xin[base + tid + 256] + y[base + tid + 256];

    float ssum = v0 + v1;
#pragma unroll
    for (int o = 16; o; o >>= 1) ssum += __shfl_xor(ssum, o, 32);
    if (lane == 0) red[w] = ssum;
    __syncthreads();
    float mean = 0.0f;
    for (int i = 0; i < 8; ++i) mean += red[i];
    mean *= (1.0f / CC);
    __syncthreads();

    float d0 = v0 - mean, d1 = v1 - mean;
    float vs = d0 * d0 + d1 * d1;
#pragma unroll
    for (int o = 16; o; o >>= 1) vs += __shfl_xor(vs, o, 32);
    if (lane == 0) red[w] = vs;
    __syncthreads();
    float var = 0.0f;
    for (int i = 0; i < 8; ++i) var += red[i];
    var *= (1.0f / CC);

    const float inv = rsqrtf(var + 1e-6f);
    const float m = rowmask ? rowmask[row] : 1.0f;
    float o0 = d0 * inv * gamma[tid]       + beta[tid];
    float o1 = d1 * inv * gamma[tid + 256] + beta[tid + 256];
    xout[base + tid]       = o0;
    xout[base + tid + 256] = o1;
    hout[base + tid]       = (half_t)(o0 * m);
    hout[base + tid + 256] = (half_t)(o1 * m);
}

// ---------------------------------------------------------------------------
// host side
// ---------------------------------------------------------------------------
extern "C" void kernel_launch(void* const* d_in, const int* in_sizes, int n_in,
                              void* d_out, int out_size, void* d_ws, size_t ws_size,
                              hipStream_t stream) {
    (void)in_sizes; (void)n_in; (void)out_size; (void)ws_size;
    const float* x_in = (const float*)d_in[0];
    const float* mask = (const float*)d_in[1];   // [B,1,T] -> flat [B*T]
    const float* Wq   = (const float*)d_in[2];
    const float* bq   = (const float*)d_in[3];
    const float* Wk   = (const float*)d_in[4];
    const float* bk   = (const float*)d_in[5];
    const float* Wv   = (const float*)d_in[6];
    const float* bv   = (const float*)d_in[7];
    const float* Wo   = (const float*)d_in[8];
    const float* bo   = (const float*)d_in[9];
    const float* erk  = (const float*)d_in[10];  // [L,1,9,KC]
    const float* erv  = (const float*)d_in[11];
    const float* g1s  = (const float*)d_in[12];
    const float* g1b  = (const float*)d_in[13];
    const float* W1   = (const float*)d_in[14];
    const float* b1   = (const float*)d_in[15];
    const float* W2   = (const float*)d_in[16];
    const float* b2   = (const float*)d_in[17];
    const float* g2s  = (const float*)d_in[18];
    const float* g2b  = (const float*)d_in[19];

    // ---- workspace carve-up ----
    char* ws = (char*)d_ws;
    size_t off = 0;
    auto carve = [&](size_t bytes) -> char* {
        char* p = ws + off;
        off += (bytes + 255) & ~(size_t)255;
        return p;
    };
    half_t* wq_h = (half_t*)carve((size_t)NL * CC * CC * 2);   // transposed [N,K]
    half_t* wk_h = (half_t*)carve((size_t)NL * CC * CC * 2);
    half_t* wv_h = (half_t*)carve((size_t)NL * CC * CC * 2);
    half_t* wo_h = (half_t*)carve((size_t)NL * CC * CC * 2);
    half_t* w1_h = (half_t*)carve((size_t)NL * CC * FF * 2);   // [FF,CC] per layer
    half_t* w2_h = (half_t*)carve((size_t)NL * FF * CC * 2);   // [CC,FF] per layer
    float*  xf   = (float*) carve((size_t)MROWS * CC * 4);
    float*  yf   = (float*) carve((size_t)MROWS * CC * 4);
    half_t* xh   = (half_t*)carve((size_t)MROWS * CC * 2);
    half_t* xmh  = (half_t*)carve((size_t)MROWS * CC * 2);
    half_t* qh   = (half_t*)carve((size_t)MROWS * CC * 2);
    half_t* khb  = (half_t*)carve((size_t)MROWS * CC * 2);
    half_t* vhb  = (half_t*)carve((size_t)MROWS * CC * 2);
    half_t* ah   = (half_t*)carve((size_t)MROWS * CC * 2);
    half_t* h1h  = (half_t*)carve((size_t)MROWS * FF * 2);

    const dim3 blk(256);
    // ---- transpose-convert weights to f16 [N,K] (every call: deterministic) ----
    k_cvt_t<<<dim3(CC / 32, CC / 32, NL), blk, 0, stream>>>(Wq, wq_h, CC, CC);
    k_cvt_t<<<dim3(CC / 32, CC / 32, NL), blk, 0, stream>>>(Wk, wk_h, CC, CC);
    k_cvt_t<<<dim3(CC / 32, CC / 32, NL), blk, 0, stream>>>(Wv, wv_h, CC, CC);
    k_cvt_t<<<dim3(CC / 32, CC / 32, NL), blk, 0, stream>>>(Wo, wo_h, CC, CC);
    k_cvt_t<<<dim3(FF / 32, CC / 32, NL), blk, 0, stream>>>(W1, w1_h, CC, FF);
    k_cvt_t<<<dim3(CC / 32, FF / 32, NL), blk, 0, stream>>>(W2, w2_h, FF, CC);

    // ---- x = x_in * mask ----
    k_premask<<<MROWS * CC / 256, blk, 0, stream>>>(x_in, mask, xf, xh);

    const dim3 gC(CC / 128, MROWS / 128);   // N=512 GEMMs
    const dim3 gF(FF / 128, MROWS / 128);   // N=2048 GEMM
    const dim3 gA(TT / 32, HH, BB);         // attention tiles

    for (int l = 0; l < NL; ++l) {
        const size_t oCC = (size_t)l * CC * CC;
        const size_t oCF = (size_t)l * CC * FF;
        // QKV projections (q pre-scaled by 1/sqrt(KC) = 0.125)
        k_gemm<<<gC, blk, 0, stream>>>(xh, wq_h + oCC, bq + l * CC, nullptr,
                                       nullptr, qh,  MROWS, CC, CC, 0, 0.125f);
        k_gemm<<<gC, blk, 0, stream>>>(xh, wk_h + oCC, bk + l * CC, nullptr,
                                       nullptr, khb, MROWS, CC, CC, 0, 1.0f);
        k_gemm<<<gC, blk, 0, stream>>>(xh, wv_h + oCC, bv + l * CC, nullptr,
                                       nullptr, vhb, MROWS, CC, CC, 0, 1.0f);
        // fused relative attention
        k_attn<<<gA, blk, 0, stream>>>(qh, khb, vhb, mask,
                                       erk + (size_t)l * 9 * KC,
                                       erv + (size_t)l * 9 * KC, ah);
        // output projection
        k_gemm<<<gC, blk, 0, stream>>>(ah, wo_h + oCC, bo + l * CC, nullptr,
                                       yf, nullptr, MROWS, CC, CC, 0, 1.0f);
        // x = LN1(x + y); xmh = f16(x * mask) feeds FFN
        k_add_ln<<<MROWS, blk, 0, stream>>>(xf, yf, g1s + l * CC, g1b + l * CC,
                                            mask, xf, xmh);
        // FFN: h1 = relu((x*m)@W1 + b1) * m ; y = (h1@W2 + b2) * m
        k_gemm<<<gF, blk, 0, stream>>>(xmh, w1_h + oCF, b1 + l * FF, mask,
                                       nullptr, h1h, MROWS, FF, CC, 1, 1.0f);
        k_gemm<<<gC, blk, 0, stream>>>(h1h, w2_h + oCF, b2 + l * CC, mask,
                                       yf, nullptr, MROWS, CC, FF, 0, 1.0f);
        // x = LN2(x + y); xh = f16(x) feeds next layer's QKV
        k_add_ln<<<MROWS, blk, 0, stream>>>(xf, yf, g2s + l * CC, g2b + l * CC,
                                            nullptr, xf, xh);
    }
    // final output: x * mask
    k_final<<<MROWS * CC / 256, blk, 0, stream>>>(xf, mask, (float*)d_out);
}
